// SecondOrderGammaDemocratic_26044681683095
// MI455X (gfx1250) — compile-verified, matrix operand order checked by reading, not run on hardware
//
#include <hip/hip_runtime.h>
#include <hip/hip_bf16.h>
#include <stdint.h>

// ---------------------------------------------------------------------------
// SecondOrderGammaDemocratic on MI455X (gfx1250, wave32, WMMA bf16)
//   K  = (Xt @ Xt^T)^2            per batch, 784x784 (padded 832), red. ch=512
//   a  <- sqrt(a+eps)/(sqrt(K a + eps)+eps)   x10   (GAMMA=0 => num==1.0f)
//   out = X diag(a) X^T           per batch, 512x512, red. N=784 (padded 800)
// GEMMs use 32x64 register macro-tiles (2x4 WMMA frags): 8 WMMA / 6 frag
// loads of reuse without exceeding the 256-VGPR window (no spills).
// ---------------------------------------------------------------------------

typedef __attribute__((ext_vector_type(16))) __bf16        v16bf;
typedef __attribute__((ext_vector_type(8)))  float         v8f;
typedef __attribute__((ext_vector_type(4)))  unsigned int  u32x4;

#define BS     32
#define CH     512
#define NTOK   784          // 28*28
#define NPAD   800          // 25*32: zero-padded reduction length, outer GEMM
#define NTOKP  832          // 13*64: zero-padded token dim for gram macro-tiles
#define EPSF   1e-10f

struct BfPack { u32x4 lo; u32x4 hi; };   // 32 bytes == v16bf

static __device__ __forceinline__ unsigned short f2bf(float f) {
  unsigned int u = __builtin_bit_cast(unsigned int, f);
  u += 0x7FFFu + ((u >> 16) & 1u);      // RNE fp32 -> bf16
  return (unsigned short)(u >> 16);
}

// Per-lane bf16 fragment for V_WMMA_F32_16X16X32_BF16 (ISA 7.12.2 layout):
// lanes 0-15: K = {0..7, 16..23}; lanes 16-31: K = {8..15, 24..31};
// lane&15 selects the row (A: M, B: N). Two contiguous 16B runs per lane.
// `elemOff` is a compile-time-constant row/k offset so the global_load_b128
// folds it into the 24-bit instruction offset (keeps pointer regs minimal).
static __device__ __forceinline__ v16bf load_frag(const unsigned short* base,
                                                  int elemOff, int lane) {
  const int kb = elemOff + ((lane & 16) ? 8 : 0);
  BfPack p;
  p.lo = *(const u32x4*)(base + kb);
  p.hi = *(const u32x4*)(base + kb + 16);
  return __builtin_bit_cast(v16bf, p);
}

// ---------------------------------------------------------------------------
// 1) Prep with LDS-tiled transpose (both sides coalesced):
//    x fp32 [BS][CH][NTOK] -> Xb bf16 [BS][CH][NPAD]  (row copy, zero pad)
//                             Xt bf16 [BS][NTOKP][CH] (transpose, zero pad)
// ---------------------------------------------------------------------------
#define NT16 (NTOKP / 16)   // 52 n-tiles
#define CT16 (CH / 16)      // 32 c-tiles

__global__ void prep_kernel(const float* __restrict__ x,
                            unsigned short* __restrict__ Xb,
                            unsigned short* __restrict__ Xt) {
  __shared__ unsigned short tile[16][17];   // +1 pad: no LDS bank conflicts
  const int bid = blockIdx.x;
  const int nt = bid % NT16;
  const int ct = (bid / NT16) % CT16;
  const int b  = bid / (NT16 * CT16);
  const int tx = threadIdx.x & 15;
  const int ty = threadIdx.x >> 4;

  // read phase: coalesced along n
  const int c = ct * 16 + ty;
  const int n = nt * 16 + tx;
  unsigned short h = 0;
  if (n < NTOK) h = f2bf(x[((size_t)b * CH + c) * NTOK + n]);
  tile[ty][tx] = h;
  if (n < NPAD) Xb[((size_t)b * CH + c) * NPAD + n] = h;
  __syncthreads();

  // write phase: coalesced along c
  const int n2 = nt * 16 + ty;              // < NTOKP always
  const int c2 = ct * 16 + tx;
  Xt[((size_t)b * NTOKP + n2) * CH + c2] = tile[tx][ty];
}

__global__ void alpha_init_kernel(float* __restrict__ a) {
  const int tid = blockIdx.x * blockDim.x + threadIdx.x;
  if (tid < BS * NTOK) a[tid] = 1.0f;
}

// ---------------------------------------------------------------------------
// 2) Token Gram: K[b] = square(Xt[b] @ Xt[b]^T), fp32 [BS][NTOKP][NTOKP]
//    one wave per 32x64 macro-tile (2x4 WMMA frags), 16 K-steps over ch=512
// ---------------------------------------------------------------------------
__global__ void gram_kernel(const unsigned short* __restrict__ Xt,
                            float* __restrict__ K) {
  const int lane = threadIdx.x & 31;
  const int wid  = blockIdx.x * 8 + (threadIdx.x >> 5);
  const int MTM = NTOKP / 32;                  // 26 row macro-tiles
  const int MTN = NTOKP / 64;                  // 13 col macro-tiles
  const int b  = wid / (MTM * MTN);
  const int t  = wid % (MTM * MTN);
  const int n0 = (t / MTN) * 32;
  const int m0 = (t % MTN) * 64;

  const unsigned short* Xtb = Xt + (size_t)b * NTOKP * CH;
  const unsigned short* aBase = Xtb + (size_t)(n0 + (lane & 15)) * CH;
  const unsigned short* bBase = Xtb + (size_t)(m0 + (lane & 15)) * CH;

  v8f acc[2][4];
#pragma unroll
  for (int i = 0; i < 2; ++i)
#pragma unroll
    for (int j = 0; j < 4; ++j) acc[i][j] = (v8f){};

  for (int k0 = 0; k0 < CH; k0 += 32) {
    v16bf af[2], bf[4];
#pragma unroll
    for (int i = 0; i < 2; ++i) af[i] = load_frag(aBase, i * 16 * CH + k0, lane);
#pragma unroll
    for (int j = 0; j < 4; ++j) bf[j] = load_frag(bBase, j * 16 * CH + k0, lane);
#pragma unroll
    for (int i = 0; i < 2; ++i)
#pragma unroll
      for (int j = 0; j < 4; ++j)
        acc[i][j] = __builtin_amdgcn_wmma_f32_16x16x32_bf16(
            false, af[i], false, bf[j], (short)0, acc[i][j], false, false);
  }

  float* Kb = K + (size_t)b * NTOKP * NTOKP;
  const int col   = lane & 15;
  const int rbase = (lane & 16) ? 8 : 0;
#pragma unroll
  for (int i = 0; i < 2; ++i)
#pragma unroll
    for (int j = 0; j < 4; ++j)
#pragma unroll
      for (int v = 0; v < 8; ++v) {
        const float g = acc[i][j][v];
        Kb[(size_t)(n0 + 16 * i + rbase + v) * NTOKP + (m0 + 16 * j + col)] = g * g;
      }
}

// ---------------------------------------------------------------------------
// 3) Sinkhorn sweep: one wave per row; K rows stream from the 192MB L2
// ---------------------------------------------------------------------------
__global__ void sinkhorn_kernel(const float* __restrict__ K,
                                const float* __restrict__ ain,
                                float* __restrict__ aout) {
  const int lane = threadIdx.x & 31;
  const int rid  = blockIdx.x * 8 + (threadIdx.x >> 5);
  const int b = rid / NTOK;
  const int r = rid % NTOK;
  const float* Krow = K + ((size_t)b * NTOKP + r) * NTOKP;
  const float* av   = ain + b * NTOK;

  float s = 0.f;
  for (int j = lane; j < NTOK; j += 32) s += Krow[j] * av[j];
  s += __shfl_xor(s, 16, 32);
  s += __shfl_xor(s, 8, 32);
  s += __shfl_xor(s, 4, 32);
  s += __shfl_xor(s, 2, 32);
  s += __shfl_xor(s, 1, 32);

  if (lane == 0) {
    const float a = av[r];
    aout[b * NTOK + r] = sqrtf(a + EPSF) / (sqrtf(s + EPSF) + EPSF);
  }
}

// ---------------------------------------------------------------------------
// 4) Xw = bf16(x * alpha) (alpha applied fp32, one side only), padded to NPAD
// ---------------------------------------------------------------------------
__global__ void weight_kernel(const float* __restrict__ x,
                              const float* __restrict__ alpha,
                              unsigned short* __restrict__ Xw) {
  size_t tid = (size_t)blockIdx.x * blockDim.x + threadIdx.x;
  if (tid >= (size_t)BS * CH * NPAD) return;
  const int n = (int)(tid % NPAD);
  const int c = (int)((tid / NPAD) % CH);
  const int b = (int)(tid / ((size_t)NPAD * CH));
  float v = 0.f;
  if (n < NTOK)
    v = x[((size_t)b * CH + c) * NTOK + n] * alpha[b * NTOK + n];
  Xw[tid] = f2bf(v);
}

// ---------------------------------------------------------------------------
// 5) Channel Gram: out[b] = Xw[b] @ Xb[b]^T, fp32 [BS][CH][CH]
//    one wave per 32x64 macro-tile, 25 K-steps over NPAD=800
// ---------------------------------------------------------------------------
__global__ void outer_kernel(const unsigned short* __restrict__ Xw,
                             const unsigned short* __restrict__ Xb,
                             float* __restrict__ out) {
  const int lane = threadIdx.x & 31;
  const int wid  = blockIdx.x * 8 + (threadIdx.x >> 5);
  const int MTM = CH / 32;                     // 16
  const int MTN = CH / 64;                     // 8
  const int b  = wid / (MTM * MTN);
  const int t  = wid % (MTM * MTN);
  const int c0 = (t / MTN) * 32;
  const int d0 = (t % MTN) * 64;

  const unsigned short* aBase = Xw + ((size_t)b * CH + c0 + (lane & 15)) * NPAD;
  const unsigned short* bBase = Xb + ((size_t)b * CH + d0 + (lane & 15)) * NPAD;

  v8f acc[2][4];
#pragma unroll
  for (int i = 0; i < 2; ++i)
#pragma unroll
    for (int j = 0; j < 4; ++j) acc[i][j] = (v8f){};

  for (int k0 = 0; k0 < NPAD; k0 += 32) {
    v16bf af[2], bf[4];
#pragma unroll
    for (int i = 0; i < 2; ++i) af[i] = load_frag(aBase, i * 16 * NPAD + k0, lane);
#pragma unroll
    for (int j = 0; j < 4; ++j) bf[j] = load_frag(bBase, j * 16 * NPAD + k0, lane);
#pragma unroll
    for (int i = 0; i < 2; ++i)
#pragma unroll
      for (int j = 0; j < 4; ++j)
        acc[i][j] = __builtin_amdgcn_wmma_f32_16x16x32_bf16(
            false, af[i], false, bf[j], (short)0, acc[i][j], false, false);
  }

  float* ob = out + (size_t)b * CH * CH;
  const int col   = lane & 15;
  const int rbase = (lane & 16) ? 8 : 0;
#pragma unroll
  for (int i = 0; i < 2; ++i)
#pragma unroll
    for (int j = 0; j < 4; ++j)
#pragma unroll
      for (int v = 0; v < 8; ++v)
        ob[(size_t)(c0 + 16 * i + rbase + v) * CH + (d0 + 16 * j + col)] =
            acc[i][j][v];
}

// ---------------------------------------------------------------------------
extern "C" void kernel_launch(void* const* d_in, const int* in_sizes, int n_in,
                              void* d_out, int out_size, void* d_ws, size_t ws_size,
                              hipStream_t stream) {
  const float* x = (const float*)d_in[0];
  float* out = (float*)d_out;

  char* ws = (char*)d_ws;
  size_t off = 0;
  auto carve = [&](size_t bytes) -> void* {
    off = (off + 255) & ~(size_t)255;
    void* p = ws + off;
    off += bytes;
    return p;
  };
  float*          K      = (float*)carve((size_t)BS * NTOKP * NTOKP * sizeof(float));
  float*          alphaA = (float*)carve((size_t)BS * NTOK * sizeof(float));
  float*          alphaB = (float*)carve((size_t)BS * NTOK * sizeof(float));
  unsigned short* Xt     = (unsigned short*)carve((size_t)BS * NTOKP * CH * 2);
  unsigned short* Xb     = (unsigned short*)carve((size_t)BS * CH * NPAD * 2);
  unsigned short* Xw     = (unsigned short*)carve((size_t)BS * CH * NPAD * 2);
  (void)ws_size; (void)in_sizes; (void)n_in; (void)out_size;

  prep_kernel<<<BS * CT16 * NT16, 256, 0, stream>>>(x, Xb, Xt);
  alpha_init_kernel<<<(BS * NTOK + 255) / 256, 256, 0, stream>>>(alphaA);

  // gram: 26x13 macro-tiles/batch, 8 waves/block -> 32*338/8 = 1352 blocks
  gram_kernel<<<BS * (NTOKP / 32) * (NTOKP / 64) / 8, 256, 0, stream>>>(Xt, K);

  float* ain = alphaA;
  float* aout = alphaB;
  for (int i = 0; i < 10; ++i) {
    sinkhorn_kernel<<<BS * NTOK / 8, 256, 0, stream>>>(K, ain, aout);
    float* tmp = ain; ain = aout; aout = tmp;
  }
  // 10 swaps (even) -> final alpha back in alphaA == ain

  weight_kernel<<<(int)(((size_t)BS * CH * NPAD + 255) / 256), 256, 0, stream>>>(
      x, ain, Xw);

  // outer: 16x8 macro-tiles/batch -> 32*128/8 = 512 blocks
  outer_kernel<<<BS * (CH / 32) * (CH / 64) / 8, 256, 0, stream>>>(Xw, Xb, out);
}
